// FeatureUpsampleModule_30064771072405
// MI455X (gfx1250) — matrix operand order repaired and checked by reading, not compiled, not run on hardware
//
#include <hip/hip_runtime.h>
#include <hip/hip_bf16.h>

// ---------------------------------------------------------------------------
// Types for CDNA5 WMMA (wave32): v_wmma_f32_16x16x32_bf16
// ---------------------------------------------------------------------------
typedef __attribute__((ext_vector_type(16))) __bf16    v16bf;
typedef __attribute__((ext_vector_type(8)))  float     v8f;
typedef __attribute__((ext_vector_type(4)))  unsigned  v4u;

union Frag16 { v16bf v; v4u u[2]; };

__device__ __forceinline__ unsigned short f2bf_bits(float f) {
  unsigned u = __float_as_uint(f);
  u += 0x7fffu + ((u >> 16) & 1u);      // round-to-nearest-even
  return (unsigned short)(u >> 16);
}
__device__ __forceinline__ __bf16 f2bf(float f) {
  unsigned short s = f2bf_bits(f);
  __bf16 r; __builtin_memcpy(&r, &s, 2); return r;
}
__device__ __forceinline__ float bf2f(__bf16 h) {
  unsigned short s; __builtin_memcpy(&s, &h, 2);
  return __uint_as_float((unsigned)s << 16);
}

static constexpr int NP = 150000;   // points (multiple of 16)

// A-fragment: lane hl=0 -> K0..15, hl=1 -> K16..31
__device__ __forceinline__ void load_a(Frag16& a, const __bf16* arow,
                                       int kk, int hl) {
  a.u[0] = *(const v4u*)(arow + kk + hl * 16);
  a.u[1] = *(const v4u*)(arow + kk + hl * 16 + 8);
}
// B-fragment from transposed weights: lane = column, contiguous 16 K values
__device__ __forceinline__ void load_b(Frag16& b, const __bf16* brow,
                                       int kk, int hl) {
  b.u[0] = *(const v4u*)(brow + kk + hl * 16);
  b.u[1] = *(const v4u*)(brow + kk + hl * 16 + 8);
}

// ---------------------------------------------------------------------------
// Gather sparse-conv GEMM + fused deterministic BN statistics.
//   Y[m,n] = sum_k sum_c X[nbr[k][m], c] * Wt[k][n][c]
// X : (NP+1) x Cin bf16 (row NP is the zero pad row)
// Wt: [8][Cout][Cin] bf16 (pre-transposed: B-frag = contiguous K per lane)
// Y : NP x Cout bf16
// P : [gridDim.x][2*Cout] fp32 per-block column partials (sum, sumsq)
// Wave tile: 16 rows x (16*NT) cols; block = 8 waves = 128 rows.
// Software-pipelined: iteration i+1 loads issued before iteration i WMMAs.
// ---------------------------------------------------------------------------
template <int NT>
__global__ __launch_bounds__(256) void conv_gemm_bf16(
    const __bf16* __restrict__ X, const int* __restrict__ nbr,
    const __bf16* __restrict__ Wt, __bf16* __restrict__ Y,
    float* __restrict__ P, int Cin, int Cout)
{
  __shared__ float Ssum[NT][8][32];
  __shared__ float Ssq [NT][8][32];

  const int lane = threadIdx.x & 31;
  const int wv   = threadIdx.x >> 5;
  const int m0   = (blockIdx.x * 8 + wv) * 16;
  const int n0   = blockIdx.y * (16 * NT);
  const int l15  = lane & 15;
  const int hl   = lane >> 4;

  v8f c[NT];
  #pragma unroll
  for (int t = 0; t < NT; ++t) c[t] = (v8f){};

  const bool active = (m0 < NP);       // wave-uniform: EXEC all-1 inside
  if (active) {
    const int row = m0 + l15;
    int srcs[8];                       // hoist gather indices: no inner dep
    #pragma unroll
    for (int k = 0; k < 8; ++k) srcs[k] = nbr[k * NP + row];

    for (int k = 0; k < 8; ++k) {
      const __bf16* arow = X + (size_t)srcs[k] * Cin;
      const __bf16* brow = Wt + ((size_t)k * Cout + n0 + l15) * Cin;
      Frag16 a0, b0[NT];
      load_a(a0, arow, 0, hl);
      #pragma unroll
      for (int t = 0; t < NT; ++t) load_b(b0[t], brow + (size_t)(16 * t) * Cin, 0, hl);

      #pragma unroll 2
      for (int kk = 0; kk < Cin; kk += 32) {
        const int nk = kk + 32;
        const int pk = (nk < Cin) ? nk : kk;   // branchless tail (cache hit)
        Frag16 a1, b1[NT];
        load_a(a1, arow, pk, hl);
        #pragma unroll
        for (int t = 0; t < NT; ++t) load_b(b1[t], brow + (size_t)(16 * t) * Cin, pk, hl);
        #pragma unroll
        for (int t = 0; t < NT; ++t)
          c[t] = __builtin_amdgcn_wmma_f32_16x16x32_bf16(
                     false, a0.v, false, b0[t].v, (short)0, c[t], false, false);
        a0 = a1;
        #pragma unroll
        for (int t = 0; t < NT; ++t) b0[t] = b1[t];
      }
    }
    // C layout: VGPR v -> row m0+v+8*hl, col n0+16*t+l15
    #pragma unroll
    for (int t = 0; t < NT; ++t) {
      #pragma unroll
      for (int v = 0; v < 8; ++v) {
        const int m = m0 + v + hl * 8;
        Y[(size_t)m * Cout + n0 + 16 * t + l15] = f2bf(c[t][v]);
      }
    }
  }

  // fused BN statistics: fold this lane's 8 rows, park in LDS
  #pragma unroll
  for (int t = 0; t < NT; ++t) {
    float s = 0.f, q = 0.f;
    #pragma unroll
    for (int v = 0; v < 8; ++v) { const float x = c[t][v]; s += x; q += x * x; }
    Ssum[t][wv][lane] = s;
    Ssq [t][wv][lane] = q;
  }
  __syncthreads();
  const int tid = threadIdx.x;
  if (tid < 16 * NT) {                 // one thread per column of this block
    const int t = tid >> 4, l = tid & 15;
    float s = 0.f, q = 0.f;
    #pragma unroll
    for (int w = 0; w < 8; ++w) {
      s += Ssum[t][w][l] + Ssum[t][w][l + 16];
      q += Ssq [t][w][l] + Ssq [t][w][l + 16];
    }
    float* prow = P + (size_t)blockIdx.x * 2 * Cout;
    const int col = n0 + tid;          // n0 + 16*t + l
    prow[col]        = s;
    prow[Cout + col] = q;
  }
}

// ---------------------------------------------------------------------------
// Dense GEMM (FC / skip): Yout = A @ W^T  (+ bias | + residual A)
// mode 0: +bias   -> bf16 Yb
// mode 1: +A[m,n] -> bf16 Yb   (skip connection, Cin == Cout)
// mode 2: +A[m,n] -> fp32 Yf   (final stage writes d_out)
// ---------------------------------------------------------------------------
template <int NT>
__global__ __launch_bounds__(256) void dense_gemm_bf16(
    const __bf16* __restrict__ A, const __bf16* __restrict__ Wt,
    const float* __restrict__ bias, __bf16* __restrict__ Yb,
    float* __restrict__ Yf, int Cin, int Cout, int mode)
{
  const int lane = threadIdx.x & 31;
  const int wv   = threadIdx.x >> 5;
  const int m0   = (blockIdx.x * 8 + wv) * 16;
  if (m0 >= NP) return;
  const int n0  = blockIdx.y * (16 * NT);
  const int l15 = lane & 15;
  const int hl  = lane >> 4;

  v8f c[NT];
  #pragma unroll
  for (int t = 0; t < NT; ++t) c[t] = (v8f){};

  const __bf16* arow = A + (size_t)(m0 + l15) * Cin;
  const __bf16* brow = Wt + (size_t)(n0 + l15) * Cin;

  Frag16 a0, b0[NT];
  load_a(a0, arow, 0, hl);
  #pragma unroll
  for (int t = 0; t < NT; ++t) load_b(b0[t], brow + (size_t)(16 * t) * Cin, 0, hl);

  #pragma unroll 2
  for (int kk = 0; kk < Cin; kk += 32) {
    const int nk = kk + 32;
    const int pk = (nk < Cin) ? nk : kk;       // branchless tail
    Frag16 a1, b1[NT];
    load_a(a1, arow, pk, hl);
    #pragma unroll
    for (int t = 0; t < NT; ++t) load_b(b1[t], brow + (size_t)(16 * t) * Cin, pk, hl);
    #pragma unroll
    for (int t = 0; t < NT; ++t)
      c[t] = __builtin_amdgcn_wmma_f32_16x16x32_bf16(
                 false, a0.v, false, b0[t].v, (short)0, c[t], false, false);
    a0 = a1;
    #pragma unroll
    for (int t = 0; t < NT; ++t) b0[t] = b1[t];
  }

  float bb[NT];
  #pragma unroll
  for (int t = 0; t < NT; ++t)
    bb[t] = (mode == 0) ? bias[n0 + 16 * t + l15] : 0.f;

  #pragma unroll
  for (int t = 0; t < NT; ++t) {
    #pragma unroll
    for (int v = 0; v < 8; ++v) {
      const int m = m0 + v + hl * 8;
      const int col = n0 + 16 * t + l15;
      float r = bb[t];
      if (mode != 0) r = bf2f(A[(size_t)m * Cin + col]);   // Cin == Cout
      const float o = c[t][v] + r;
      if (mode == 2) Yf[(size_t)m * Cout + col] = o;
      else           Yb[(size_t)m * Cout + col] = f2bf(o);
    }
  }
}

// ---------------------------------------------------------------------------
// Helpers: weight transpose/pack, fp32->bf16, pad-row zero, BN finalize/apply
// ---------------------------------------------------------------------------
__global__ void pack_t_kernel(const float* __restrict__ W, __bf16* __restrict__ Wt,
                              int K, int Cin, int Cout)
{ // W: [K][Cin][Cout] fp32 -> Wt: [K][Cout][Cin] bf16
  const size_t total = (size_t)K * Cin * Cout;
  for (size_t e = (size_t)blockIdx.x * blockDim.x + threadIdx.x; e < total;
       e += (size_t)gridDim.x * blockDim.x) {
    const size_t kk = e / ((size_t)Cin * Cout);
    const size_t r  = e - kk * (size_t)Cin * Cout;
    const size_t ci = r / Cout, co = r % Cout;
    Wt[(kk * Cout + co) * Cin + ci] = f2bf(W[e]);
  }
}

__global__ void f32_to_bf16_kernel(const float* __restrict__ s,
                                   __bf16* __restrict__ d, size_t n) {
  for (size_t e = (size_t)blockIdx.x * blockDim.x + threadIdx.x; e < n;
       e += (size_t)gridDim.x * blockDim.x) d[e] = f2bf(s[e]);
}

__global__ void zero_bf16_kernel(__bf16* p, int n) {
  int i = blockIdx.x * blockDim.x + threadIdx.x;
  if (i < n) p[i] = f2bf(0.f);
}

__global__ void bn_finalize_kernel(const float* __restrict__ P,
                                   const float* __restrict__ g,
                                   const float* __restrict__ b,
                                   float* __restrict__ ab, int Cout, int nblk) {
  const int c = blockIdx.x * blockDim.x + threadIdx.x;
  if (c >= Cout) return;
  float s = 0.f, q = 0.f;
  for (int p = 0; p < nblk; ++p) {
    s += P[(size_t)p * 2 * Cout + c];
    q += P[(size_t)p * 2 * Cout + Cout + c];
  }
  const float mean = s / (float)NP;
  const float var  = q / (float)NP - mean * mean;
  const float sc   = g[c] * __frsqrt_rn(var + 1e-5f);
  ab[c]        = sc;
  ab[Cout + c] = b[c] - mean * sc;
}

__global__ void bn_relu_kernel(__bf16* __restrict__ Y,
                               const float* __restrict__ ab,
                               size_t total, int Cout) {
  for (size_t e = (size_t)blockIdx.x * blockDim.x + threadIdx.x; e < total;
       e += (size_t)gridDim.x * blockDim.x) {
    const int c = (int)(e % (size_t)Cout);
    float v = bf2f(Y[e]);
    v = fmaf(ab[c], v, ab[Cout + c]);
    Y[e] = f2bf(fmaxf(v, 0.f));
  }
}

// ---------------------------------------------------------------------------
// Orchestration
// ---------------------------------------------------------------------------
extern "C" void kernel_launch(void* const* d_in, const int* in_sizes, int n_in,
                              void* d_out, int out_size, void* d_ws, size_t ws_size,
                              hipStream_t stream) {
  (void)in_sizes; (void)n_in; (void)out_size; (void)ws_size;
  const float* feats = (const float*)d_in[0];
  const int*   nbr   = (const int*)d_in[1];
  const float* fc_w  = (const float*)d_in[2];
  const float* fc_b  = (const float*)d_in[3];
  const float* conv_w[4] = {(const float*)d_in[4],  (const float*)d_in[8],
                            (const float*)d_in[12], (const float*)d_in[16]};
  const float* bn_g[4]   = {(const float*)d_in[5],  (const float*)d_in[9],
                            (const float*)d_in[13], (const float*)d_in[17]};
  const float* bn_b[4]   = {(const float*)d_in[6],  (const float*)d_in[10],
                            (const float*)d_in[14], (const float*)d_in[18]};
  const float* skip_w[4] = {(const float*)d_in[7],  (const float*)d_in[11],
                            (const float*)d_in[15], (const float*)d_in[19]};

  const int Co[4]   = {256, 512, 256, 96};
  const int CinS[4] = {256, 256, 512, 256};

  // --- workspace carve-up (256B aligned) ---
  char* base = (char*)d_ws;
  size_t off = 0;
  auto carve = [&](size_t bytes) -> char* {
    char* p = base + off;
    off += (bytes + 255) & ~(size_t)255;
    return p;
  };
  __bf16* XA = (__bf16*)carve((size_t)(NP + 1) * 512 * 2);
  __bf16* XB = (__bf16*)carve((size_t)(NP + 1) * 512 * 2);
  __bf16* Yb = (__bf16*)carve((size_t)NP * 512 * 2);
  __bf16* cwT[4];
  for (int i = 0; i < 4; ++i) cwT[i] = (__bf16*)carve((size_t)8 * CinS[i] * Co[i] * 2);
  __bf16* fcT = (__bf16*)carve((size_t)256 * 96 * 2);
  __bf16* swT[4];
  for (int i = 0; i < 4; ++i) swT[i] = (__bf16*)carve((size_t)Co[i] * Co[i] * 2);
  const int GX = ((NP / 16) + 7) / 8;           // 1172 blocks (8 waves x 16 rows)
  float* P  = (float*)carve((size_t)GX * 2 * 512 * 4);
  float* ab = (float*)carve((size_t)2 * 512 * 4);

  // --- pack weights to bf16 transposed layouts ---
  pack_t_kernel<<<2048, 256, 0, stream>>>(fc_w, fcT, 1, 96, 256);
  for (int i = 0; i < 4; ++i) {
    pack_t_kernel<<<2048, 256, 0, stream>>>(conv_w[i], cwT[i], 8, CinS[i], Co[i]);
    pack_t_kernel<<<2048, 256, 0, stream>>>(skip_w[i], swT[i], 1, Co[i], Co[i]);
  }

  // --- feats -> bf16, FC layer -> XB, zero pad row ---
  f32_to_bf16_kernel<<<4096, 256, 0, stream>>>(feats, XA, (size_t)NP * 96);
  dense_gemm_bf16<4><<<dim3(GX, 256 / 64), 256, 0, stream>>>(
      XA, fcT, fc_b, XB, nullptr, 96, 256, 0);
  zero_bf16_kernel<<<1, 512, 0, stream>>>(XB + (size_t)NP * 256, 256);

  // --- four residual sparse-conv stages ---
  __bf16* cur = XB; __bf16* oth = XA;
  for (int i = 0; i < 4; ++i) {
    const int Ci = CinS[i], C = Co[i];
    if (C % 64 == 0)
      conv_gemm_bf16<4><<<dim3(GX, C / 64), 256, 0, stream>>>(
          cur, nbr, cwT[i], Yb, P, Ci, C);
    else
      conv_gemm_bf16<2><<<dim3(GX, C / 32), 256, 0, stream>>>(
          cur, nbr, cwT[i], Yb, P, Ci, C);
    bn_finalize_kernel<<<(C + 255) / 256, 256, 0, stream>>>(
        P, bn_g[i], bn_b[i], ab, C, GX);
    bn_relu_kernel<<<4096, 256, 0, stream>>>(Yb, ab, (size_t)NP * C, C);
    if (i < 3) {
      dense_gemm_bf16<4><<<dim3(GX, C / 64), 256, 0, stream>>>(
          Yb, swT[i], nullptr, oth, nullptr, C, C, 1);
      zero_bf16_kernel<<<1, 512, 0, stream>>>(oth + (size_t)NP * C, C);
      __bf16* t = cur; cur = oth; oth = t;
    } else {
      dense_gemm_bf16<2><<<dim3(GX, C / 32), 256, 0, stream>>>(
          Yb, swT[i], nullptr, nullptr, (float*)d_out, C, C, 2);
    }
  }
}
